// CINComp_2783138808385
// MI455X (gfx1250) — compile-verified, matrix operand order checked by reading, not run on hardware
//
#include <hip/hip_runtime.h>

// Shapes (fixed by the reference): B=1024, HK=64, H0=64, D=32, O=128
#define NB    1024
#define HK    64
#define H0DIM 64
#define DDIM  32
#define ODIM  128
#define KTOT  (HK * H0DIM)   // 4096
#define KC    256            // K elements staged in LDS per chunk (4 i-rows)
#define NBPB  2              // batches per block (share the W chunk)

#define WSTRIDE 264          // padded LDS row stride (halves) for W chunk (528B: 16B-aligned, bank-rotating)
#define FSTRIDE 72           // padded LDS row stride (halves) for feat/base (144B: 16B-aligned)

#define SW_HALVES  (ODIM * WSTRIDE)             // 33792
#define SFB_HALVES (DDIM * FSTRIDE)             // 2304 per batch per array
#define LDS_HALVES (SW_HALVES + 2 * NBPB * SFB_HALVES)  // 43008
#define LDS_BYTES  (LDS_HALVES * 2)             // 86016 bytes

typedef __attribute__((ext_vector_type(4)))  _Float16 v4h;
typedef __attribute__((ext_vector_type(8)))  _Float16 v8h;
typedef __attribute__((ext_vector_type(16))) _Float16 v16h;
typedef __attribute__((ext_vector_type(4)))  float    v4f;
typedef __attribute__((ext_vector_type(8)))  float    v8f;

__global__ __launch_bounds__(256)
void cin_wmma_f16_kernel(const float* __restrict__ feat,
                         const float* __restrict__ basep,
                         const float* __restrict__ W,
                         const float* __restrict__ bias,
                         float* __restrict__ out)
{
    extern __shared__ __align__(16) _Float16 smem[];
    _Float16* sW = smem;                          // [128][WSTRIDE] f16 W chunk (shared by both batches)
    _Float16* sF = smem + SW_HALVES;              // [NBPB][32][FSTRIDE] feature, transposed [d][i]
    _Float16* sB = sF + NBPB * SFB_HALVES;        // [NBPB][32][FSTRIDE] base,    transposed [d][j]

    const int tid = threadIdx.x;                  // 256 threads = 8 waves (wave32)
    const int b0  = blockIdx.x * NBPB;

    // ---- stage feature/base for both batches into LDS, f32->f16, transposed ----
    {
        const float* fp = feat  + (size_t)b0 * (HK * DDIM);     // contiguous across the 2 batches
        const float* bp = basep + (size_t)b0 * (HK * DDIM);
        for (int n = tid; n < NBPB * HK * DDIM; n += 256) {     // 4096 elems, coalesced
            int bl = n >> 11;        // local batch 0..1
            int r  = n & 2047;
            int i  = r >> 5;         // 0..63
            int d  = r & 31;         // 0..31
            sF[bl * SFB_HALVES + d * FSTRIDE + i] = (_Float16)fp[n];
            sB[bl * SFB_HALVES + d * FSTRIDE + i] = (_Float16)bp[n];
        }
    }

    const int wave   = tid >> 5;
    const int lane   = tid & 31;
    const int nh     = wave & 1;             // d-half: columns [nh*16, nh*16+16)
    const int mh     = (wave >> 1) & 1;      // row-half: rows [mh*64, mh*64+64) = 4 M-tiles
    const int bh     = wave >> 2;            // local batch 0..1
    const int lane16 = lane & 15;
    const int kb     = (lane >> 4) << 3;     // 0 or 8 (16-bit WMMA K sub-block per lane half)
    const int d      = nh * 16 + lane16;     // this lane's N index (= d)

    const _Float16* sFb = sF + bh * SFB_HALVES;
    const _Float16* sBb = sB + bh * SFB_HALVES;

    v8f acc[4] = {};   // 4 M-tiles of 16 rows each: rows mh*64 + mt*16 ..

    const int wrow  = tid >> 1;              // W staging: row 0..127
    const int wcolb = (tid & 1) * (KC / 2);  // 0 or 128

    for (int kk = 0; kk < KTOT / KC; ++kk) {  // 16 K-chunks
        __syncthreads();
        // ---- cooperative stage of W[:, kk*KC .. +KC) as f16 into LDS (once per 2 batches) ----
        {
            const float* wp = W  + (size_t)wrow * KTOT + kk * KC + wcolb;
            _Float16*    dp = sW + wrow * WSTRIDE + wcolb;
            #pragma unroll
            for (int c = 0; c < KC / 2; c += 4) {
                v4f x = *(const v4f*)(wp + c);                       // global_load_b128
                v4h h = { (_Float16)x.x, (_Float16)x.y,
                          (_Float16)x.z, (_Float16)x.w };
                *(v4h*)(dp + c) = h;                                 // ds_store_b64
            }
        }
        __syncthreads();

        #pragma unroll
        for (int il = 0; il < KC / 64; ++il) {        // 4 i-rows per chunk
            const int i = kk * (KC / 64) + il;
            const _Float16 f = sFb[d * FSTRIDE + i];  // feature[b,i,d] broadcast per lane
            #pragma unroll
            for (int jj = 0; jj < 2; ++jj) {          // 32-wide j sub-chunk => WMMA K=32
                // B fragment synthesized in registers: B[k,n] = f * base[b,j,d]
                const int jb = jj * 32 + kb;
                v8h b0v = *(const v8h*)(sBb + d * FSTRIDE + jb);        // ds_load_b128
                v8h b1v = *(const v8h*)(sBb + d * FSTRIDE + jb + 16);   // ds_load_b128
                v16h bf;
                #pragma unroll
                for (int q = 0; q < 8; ++q) { bf[q] = b0v[q] * f; bf[q + 8] = b1v[q] * f; }

                const int c0 = il * 64 + jj * 32 + kb;  // chunk-local K offset
                // Reuse B fragment across 4 M-tiles (4 independent WMMA chains)
                #pragma unroll
                for (int mt = 0; mt < 4; ++mt) {
                    const _Float16* ap = sW + (mh * 64 + mt * 16 + lane16) * WSTRIDE + c0;
                    v8h a0 = *(const v8h*)(ap);
                    v8h a1 = *(const v8h*)(ap + 16);
                    v16h af = __builtin_shufflevector(a0, a1,
                                0,1,2,3,4,5,6,7,8,9,10,11,12,13,14,15);
                    acc[mt] = __builtin_amdgcn_wmma_f32_16x16x32_f16(
                                false, af, false, bf, (short)0, acc[mt], false, false);
                }
            }
        }
    }

    // ---- epilogue: f32 C layout -> out[b,o,d] + bias[o] ----
    // VGPR r: lanes 0-15 hold M=r, lanes 16-31 hold M=8+r; N = lane&15.
    const int rbase = (lane < 16) ? 0 : 8;
    float* op = out + (size_t)(b0 + bh) * (ODIM * DDIM);
    #pragma unroll
    for (int mt = 0; mt < 4; ++mt) {
        #pragma unroll
        for (int r = 0; r < 8; ++r) {
            int o = mh * 64 + mt * 16 + rbase + r;
            op[o * DDIM + d] = acc[mt][r] + bias[o];
        }
    }
}

extern "C" void kernel_launch(void* const* d_in, const int* in_sizes, int n_in,
                              void* d_out, int out_size, void* d_ws, size_t ws_size,
                              hipStream_t stream) {
    (void)in_sizes; (void)n_in; (void)d_ws; (void)ws_size; (void)out_size;
    const float* feat  = (const float*)d_in[0];  // (1024, 64, 32) f32
    const float* basep = (const float*)d_in[1];  // (1024, 64, 32) f32
    const float* W     = (const float*)d_in[2];  // (128, 4096)   f32
    const float* bias  = (const float*)d_in[3];  // (128,)        f32
    float* out = (float*)d_out;                  // (1024, 128, 32) f32

    dim3 grid(NB / NBPB), block(256);
    cin_wmma_f16_kernel<<<grid, block, LDS_BYTES, stream>>>(feat, basep, W, bias, out);
}
